// LocalAttention_45037027065913
// MI455X (gfx1250) — matrix operand order repaired and checked by reading, not compiled, not run on hardware
//
#include <hip/hip_runtime.h>
#include <hip/hip_bf16.h>
#include <math.h>

typedef __attribute__((ext_vector_type(16))) _Float16 v16h;
typedef __attribute__((ext_vector_type(8)))  _Float16 v8h;
typedef __attribute__((ext_vector_type(8)))  float    v8f;

#define B_    256
#define T_    2048
#define E_    128
#define WIN_  5
#define OC_   128
#define HALO  2
#define TROWS (16 + 2*HALO)   // 20 rows: 16-row tile + window halo
#define EP_   (E_ + 4)        // padded f32 LDS row stride (floats)
#define GP_   (E_ + 8)        // padded f16 gated-tile row stride (halfs), 272B = 4-bank rot

union Frag16  { v16h v; v8h half[2]; _Float16 h[16]; };

// One block: batch b, contiguous t-chunk of T_/splits positions.
// 8 waves; wave w owns output columns [16w, 16w+16).
// Per 16-row t-tile:
//   gather 20 emb rows -> LDS (f32)  ->  scores (sigmoid)  ->
//   gate+convert ONCE to f16 LDS tile  ->  all 8 waves ds_load A frags -> 4x WMMA.
__global__ __launch_bounds__(256)
void local_attn_main(const int*   __restrict__ x,
                     const float* __restrict__ emb,
                     const float* __restrict__ attw,
                     const float* __restrict__ attb,
                     const float* __restrict__ cnnw,
                     const float* __restrict__ cnnb,
                     float*       __restrict__ part,
                     int splits, int apply_tanh)
{
    __shared__ __align__(16) float    emb_s[TROWS * EP_];
    __shared__ __align__(16) _Float16 gated_s[16 * GP_];
    __shared__ float attw_s[WIN_ * E_];
    __shared__ float sc_s[16];

    const int tid    = threadIdx.x;
    const int wave   = tid >> 5;
    const int lane   = tid & 31;
    const int nloc   = lane & 15;     // N column within wave tile / M row for A
    const int hi     = lane >> 4;     // half selector in WMMA layouts
    const int b      = blockIdx.x;
    const int split  = blockIdx.y;
    const int tchunk = T_ / splits;
    const int t0     = split * tchunk;
    const int ntiles = tchunk >> 4;

    // att_w -> LDS (once per block)
    for (int i = tid; i < WIN_ * E_; i += 256) attw_s[i] = attw[i];
    const float attb0 = attb[0];

    // B fragments: B[k][n] = cnn_w[n][k]; lane%16 = N, K = 32c + 8*hi + e (e<8), 32c + 16 + 8*hi + (e-8) (e>=8)
    // -> per lane two 8-float runs of a cnn_w row, starting 16 floats apart. Loaded once.
    const int oc = (wave << 4) + nloc;
    Frag16 bfrag[4];
    #pragma unroll
    for (int c = 0; c < 4; ++c) {
        const float* src = cnnw + (size_t)oc * E_ + c * 32 + hi * 8;
        #pragma unroll
        for (int e = 0; e < 16; ++e) bfrag[c].h[e] = (_Float16)src[e + (e >= 8 ? 8 : 0)];
    }
    const float bias = cnnb[oc];

    float runmax = -1e30f;   // max of pre-activation (tanh applied once at the end)

    for (int tile = 0; tile < ntiles; ++tile) {
        const int tbase = t0 + (tile << 4);
        __syncthreads();  // previous tile's LDS consumers are done

        // Gather 20 rows x 128 f32 (float4 granularity); zero-pad halo outside [0,T)
        for (int idx = tid; idx < TROWS * (E_ / 4); idx += 256) {
            const int r  = idx >> 5;
            const int c4 = idx & 31;
            const int tg = tbase - HALO + r;
            float4 val = make_float4(0.f, 0.f, 0.f, 0.f);
            if (tg >= 0 && tg < T_) {
                const int tok = x[(size_t)b * T_ + tg];
                val = *reinterpret_cast<const float4*>(emb + (size_t)tok * E_ + c4 * 4);
            }
            *reinterpret_cast<float4*>(&emb_s[r * EP_ + c4 * 4]) = val;
        }
        __syncthreads();

        // Scores: 16 threads per t-row reduce over E, then shfl_xor tree (16-group)
        {
            const int m  = tid >> 4;
            const int le = tid & 15;
            float p = 0.f;
            #pragma unroll
            for (int k = 0; k < WIN_; ++k) {
                const float* er = &emb_s[(m + k) * EP_];
                const float* wr = &attw_s[k * E_];
                for (int e = le; e < E_; e += 16) p += er[e] * wr[e];
            }
            p += __shfl_xor(p, 1);
            p += __shfl_xor(p, 2);
            p += __shfl_xor(p, 4);
            p += __shfl_xor(p, 8);
            if (le == 0) sc_s[m] = 1.f / (1.f + expf(-(p + attb0)));
        }
        __syncthreads();

        // Gate + f16 convert ONCE (shared by all 8 waves):
        // thread j: row m = j>>4, 8 contiguous elements -> one 16B LDS store
        {
            const int m  = tid >> 4;
            const int e0 = (tid & 15) * 8;
            const float sm = sc_s[m];
            const float* er = &emb_s[(m + HALO) * EP_ + e0];
            v8h g;
            #pragma unroll
            for (int e = 0; e < 8; ++e) g[e] = (_Float16)(er[e] * sm);
            *reinterpret_cast<v8h*>(&gated_s[m * GP_ + e0]) = g;
        }
        __syncthreads();

        // A fragments: pure b128 LDS loads. Per K-chunk c, lane (m=nloc, hi):
        //   h[0..7]  = gated[m][32c + 8*hi      .. +8)
        //   h[8..15] = gated[m][32c + 16 + 8*hi .. +8)
        const _Float16* arow = &gated_s[nloc * GP_];
        v8f acc = {};
        #pragma unroll
        for (int c = 0; c < 4; ++c) {
            Frag16 a;
            a.half[0] = *reinterpret_cast<const v8h*>(&arow[c * 32 + 8 * hi]);
            a.half[1] = *reinterpret_cast<const v8h*>(&arow[c * 32 + 16 + 8 * hi]);
            acc = __builtin_amdgcn_wmma_f32_16x16x32_f16(
                false, a.v, false, bfrag[c].v, (short)0, acc, false, false);
        }

        // Running max of pre-activations; lane's 8 values share column oc
        #pragma unroll
        for (int r = 0; r < 8; ++r)
            runmax = fmaxf(runmax, acc[r] + bias);
    }

    // Merge the two M-halves: lanes L and L+16 hold the same output column
    runmax = fmaxf(runmax, __shfl_xor(runmax, 16));
    if (apply_tanh) runmax = tanhf(runmax);
    if (lane < 16)
        part[((size_t)b * splits + split) * OC_ + (wave << 4) + lane] = runmax;
}

// Deterministic cross-split max; tanh applied here (monotone => commutes with max)
__global__ void reduce_splits(const float* __restrict__ part,
                              float* __restrict__ out, int splits)
{
    const int i = blockIdx.x * blockDim.x + threadIdx.x;
    if (i >= B_ * OC_) return;
    const int b = i / OC_, o = i % OC_;
    float m = -1e30f;
    for (int s = 0; s < splits; ++s)
        m = fmaxf(m, part[((size_t)b * splits + s) * OC_ + o]);
    out[i] = tanhf(m);
}

extern "C" void kernel_launch(void* const* d_in, const int* in_sizes, int n_in,
                              void* d_out, int out_size, void* d_ws, size_t ws_size,
                              hipStream_t stream) {
    const int*   x    = (const int*)  d_in[0];
    const float* emb  = (const float*)d_in[1];
    const float* attw = (const float*)d_in[2];
    const float* attb = (const float*)d_in[3];
    const float* cnnw = (const float*)d_in[4];
    const float* cnnb = (const float*)d_in[5];
    float* out = (float*)d_out;

    int splits = 8;
    while (splits > 1 &&
           ws_size < (size_t)B_ * splits * OC_ * sizeof(float))
        splits >>= 1;

    float* part = (splits > 1) ? (float*)d_ws : out;
    const int apply_tanh_in_main = (splits == 1) ? 1 : 0;

    dim3 grid(B_, splits);
    local_attn_main<<<grid, 256, 0, stream>>>(x, emb, attw, attb, cnnw, cnnb,
                                              part, splits, apply_tanh_in_main);
    if (splits > 1)
        reduce_splits<<<(B_ * OC_ + 255) / 256, 256, 0, stream>>>(part, out, splits);
}
